// EdgeModel_83562883711137
// MI455X (gfx1250) — compile-verified, matrix operand order checked by reading, not run on hardware
//
#include <hip/hip_runtime.h>
#include <stdint.h>

typedef __attribute__((ext_vector_type(16))) __bf16 v16bf;
typedef __attribute__((ext_vector_type(8)))  float  v8f;

#define HS   256
#define CIN  384
#define KT1  12          // 384/32 k-tiles for layer 1
#define KT2  8           // 256/32 k-tiles for layer 2

union FragA { v16bf v; __bf16 h[16]; unsigned u[8]; };
union FragB { v16bf v; uint4 q[2]; };

// k-within-tile for A/B fragment VGPR v and lane-half h (16-bit WMMA layout)
__device__ __host__ __forceinline__ int frag_klocal(int v, int half) {
  return ((v < 4) ? 0 : 16) + half * 8 + 2 * (v & 3);
}

// ---------------- kernel 1: pack weights to bf16 fragment layout + zero stats ----------
__global__ void pack_weights_k(const float* __restrict__ W1, const float* __restrict__ W2,
                               __bf16* __restrict__ Wp1, __bf16* __restrict__ Wp2,
                               float* __restrict__ stats /* 1024 floats zeroed */) {
  int t = blockIdx.x * blockDim.x + threadIdx.x;
  if (t < 1024) stats[t] = 0.f;
  // layout: [n][kt][half][p(0..15)], p=2v+odd -> klocal
  if (t < HS * KT1 * 32) {
    int n = t / (KT1 * 32);
    int rem = t % (KT1 * 32);
    int kt = rem / 32, r2 = rem % 32;
    int half = r2 >> 4, p = r2 & 15;
    int k = kt * 32 + frag_klocal(p >> 1, half) + (p & 1);
    Wp1[t] = (__bf16)W1[k * HS + n];
  }
  if (t < HS * KT2 * 32) {
    int n = t / (KT2 * 32);
    int rem = t % (KT2 * 32);
    int kt = rem / 32, r2 = rem % 32;
    int half = r2 >> 4, p = r2 & 15;
    int k = kt * 32 + frag_klocal(p >> 1, half) + (p & 1);
    Wp2[t] = (__bf16)W2[k * HS + n];
  }
}

// ---------------- kernel 2: GEMM1  h = concat(src,dst,ea,u[batch]) @ W1 + b1 ----------
__global__ __launch_bounds__(256)
void gemm1_k(const float* __restrict__ srcp, const float* __restrict__ dstp,
             const float* __restrict__ eap,  const float* __restrict__ up,
             const int* __restrict__ batch,
             const __bf16* __restrict__ Wp1, const float* __restrict__ b1,
             float* __restrict__ hout,  // E x 256 (lives in d_out)
             float* __restrict__ gsum, float* __restrict__ gsq, int E)
{
  __shared__ float ssum[HS], ssq[HS];
  if (threadIdx.x < HS) { ssum[threadIdx.x] = 0.f; ssq[threadIdx.x] = 0.f; }
  __syncthreads();

  const int lane = threadIdx.x & 31;
  const int wave = threadIdx.x >> 5;
  const int msub = wave & 3;        // 4 row sub-tiles of 16
  const int nhalf = wave >> 2;      // 2 column halves of 128
  const int half = lane >> 4;
  const int ln = lane & 15;

  const int rowBase = blockIdx.x * 64 + msub * 16;
  int row = rowBase + ln;
  int rowc = row < E ? row : E - 1;
  const int colBase = nhalf * 128;
  const int bidx = batch[rowc];

  v8f acc[8] = {};

  for (int kt = 0; kt < KT1; ++kt) {
    const float* abase;
    if (kt < 4)       abase = srcp + (size_t)rowc * 128 + kt * 32;
    else if (kt < 8)  abase = dstp + (size_t)rowc * 128 + (kt - 4) * 32;
    else if (kt < 10) abase = eap  + (size_t)rowc * 64  + (kt - 8) * 32;
    else              abase = up   + (size_t)bidx * 64  + (kt - 10) * 32;

    FragA A;
    #pragma unroll
    for (int v = 0; v < 8; ++v) {
      float2 f = *(const float2*)(abase + frag_klocal(v, half));
      A.h[2 * v]     = (__bf16)f.x;   // native f32->bf16 cvt (hardware pack path)
      A.h[2 * v + 1] = (__bf16)f.y;
    }
    #pragma unroll
    for (int nt = 0; nt < 8; ++nt) {
      int n = colBase + nt * 16 + ln;
      const uint4* bp = (const uint4*)(Wp1 + ((size_t)n * KT1 + kt) * 32 + half * 16);
      FragB B; B.q[0] = bp[0]; B.q[1] = bp[1];
      acc[nt] = __builtin_amdgcn_wmma_f32_16x16x32_bf16(
          false, A.v, false, B.v, (short)0, acc[nt], false, false);
    }
  }

  const int rvalid = E - rowBase;   // #valid rows among the 16
  #pragma unroll
  for (int nt = 0; nt < 8; ++nt) {
    int n = colBase + nt * 16 + ln;
    float bias = b1[n];
    float s = 0.f, q = 0.f;
    #pragma unroll
    for (int j = 0; j < 8; ++j) {
      int r = j + half * 8;               // C layout: VGPR j -> M=j (+8 for hi lanes)
      float y = acc[nt][j] + bias;
      if (r < rvalid) {
        hout[(size_t)(rowBase + r) * HS + n] = y;
        s += y; q += y * y;
      }
    }
    s += __shfl_xor(s, 16, 32);
    q += __shfl_xor(q, 16, 32);
    if (half == 0) { atomicAdd(&ssum[n], s); atomicAdd(&ssq[n], q); }
  }
  __syncthreads();
  if (threadIdx.x < HS) {
    atomicAdd(&gsum[threadIdx.x], ssum[threadIdx.x]);
    atomicAdd(&gsq[threadIdx.x],  ssq[threadIdx.x]);
  }
}

// ---------------- kernel 3: BN stats -> per-channel scale/shift ----------
__global__ void stats_k(const float* __restrict__ gsum, const float* __restrict__ gsq,
                        const float* __restrict__ gamma, const float* __restrict__ beta,
                        float* __restrict__ scale, float* __restrict__ shift, float invE)
{
  int n = threadIdx.x;
  float mean = gsum[n] * invE;
  float var  = gsq[n] * invE - mean * mean;
  float inv  = rsqrtf(var + 1e-5f);
  float sc   = gamma[n] * inv;
  scale[n] = sc;
  shift[n] = beta[n] - mean * sc;
}

// ---------------- kernel 4: GEMM2  y = relu(BN1(h)) @ W2 + b2 (in place in d_out) ----
__global__ __launch_bounds__(256)
void gemm2_k(float* __restrict__ hio,   // d_out: read h, write y
             const __bf16* __restrict__ Wp2, const float* __restrict__ b2,
             const float* __restrict__ scale1, const float* __restrict__ shift1,
             float* __restrict__ gsum, float* __restrict__ gsq, int E)
{
  __shared__ float ssum[HS], ssq[HS], ssc[HS], ssh[HS];
  if (threadIdx.x < HS) {
    ssum[threadIdx.x] = 0.f; ssq[threadIdx.x] = 0.f;
    ssc[threadIdx.x] = scale1[threadIdx.x];
    ssh[threadIdx.x] = shift1[threadIdx.x];
  }
  __syncthreads();

  const int lane = threadIdx.x & 31;
  const int wave = threadIdx.x >> 5;
  const int msub = wave & 3;
  const int nhalf = wave >> 2;
  const int half = lane >> 4;
  const int ln = lane & 15;

  const int rowBase = blockIdx.x * 64 + msub * 16;
  int row = rowBase + ln;
  int rowc = row < E ? row : E - 1;
  const int colBase = nhalf * 128;

  v8f acc[8] = {};

  for (int kt = 0; kt < KT2; ++kt) {
    const float* abase = hio + (size_t)rowc * HS + kt * 32;
    FragA A;
    #pragma unroll
    for (int v = 0; v < 8; ++v) {
      int kl = frag_klocal(v, half);
      int k = kt * 32 + kl;
      float2 f = *(const float2*)(abase + kl);
      A.h[2 * v]     = (__bf16)fmaxf(0.f, f.x * ssc[k]     + ssh[k]);
      A.h[2 * v + 1] = (__bf16)fmaxf(0.f, f.y * ssc[k + 1] + ssh[k + 1]);
    }
    #pragma unroll
    for (int nt = 0; nt < 8; ++nt) {
      int n = colBase + nt * 16 + ln;
      const uint4* bp = (const uint4*)(Wp2 + ((size_t)n * KT2 + kt) * 32 + half * 16);
      FragB B; B.q[0] = bp[0]; B.q[1] = bp[1];
      acc[nt] = __builtin_amdgcn_wmma_f32_16x16x32_bf16(
          false, A.v, false, B.v, (short)0, acc[nt], false, false);
    }
  }

  // both N-half waves read the same h rows this block is about to overwrite:
  // all reads must retire before any epilogue store.
  __syncthreads();

  const int rvalid = E - rowBase;
  #pragma unroll
  for (int nt = 0; nt < 8; ++nt) {
    int n = colBase + nt * 16 + ln;
    float bias = b2[n];
    float s = 0.f, q = 0.f;
    #pragma unroll
    for (int j = 0; j < 8; ++j) {
      int r = j + half * 8;
      float y = acc[nt][j] + bias;
      if (r < rvalid) {
        hio[(size_t)(rowBase + r) * HS + n] = y;
        s += y; q += y * y;
      }
    }
    s += __shfl_xor(s, 16, 32);
    q += __shfl_xor(q, 16, 32);
    if (half == 0) { atomicAdd(&ssum[n], s); atomicAdd(&ssq[n], q); }
  }
  __syncthreads();
  if (threadIdx.x < HS) {
    atomicAdd(&gsum[threadIdx.x], ssum[threadIdx.x]);
    atomicAdd(&gsq[threadIdx.x],  ssq[threadIdx.x]);
  }
}

// ---------------- kernel 5: final BN2 affine, in place, float4 ----------
__global__ __launch_bounds__(256)
void bn2_k(float* __restrict__ out, const float* __restrict__ scale,
           const float* __restrict__ shift, size_t total)
{
  size_t i = (size_t)blockIdx.x * blockDim.x + threadIdx.x;
  size_t idx4 = i * 4;
  if (idx4 < total) {
    float4 v = *(float4*)(out + idx4);
    int n = (int)(idx4 & (HS - 1));
    v.x = v.x * scale[n]     + shift[n];
    v.y = v.y * scale[n + 1] + shift[n + 1];
    v.z = v.z * scale[n + 2] + shift[n + 2];
    v.w = v.w * scale[n + 3] + shift[n + 3];
    *(float4*)(out + idx4) = v;
  }
}

extern "C" void kernel_launch(void* const* d_in, const int* in_sizes, int n_in,
                              void* d_out, int out_size, void* d_ws, size_t ws_size,
                              hipStream_t stream) {
  const float* srcp = (const float*)d_in[0];
  const float* dstp = (const float*)d_in[1];
  const float* eap  = (const float*)d_in[2];
  const float* up   = (const float*)d_in[3];
  const float* W1   = (const float*)d_in[4];
  const float* b1   = (const float*)d_in[5];
  const float* g1   = (const float*)d_in[6];
  const float* be1  = (const float*)d_in[7];
  const float* W2   = (const float*)d_in[8];
  const float* b2   = (const float*)d_in[9];
  const float* g2   = (const float*)d_in[10];
  const float* be2  = (const float*)d_in[11];
  const int*   batch= (const int*)d_in[12];
  float* out = (float*)d_out;

  const int E = in_sizes[0] / 128;

  // workspace layout (all well under 1 MB)
  __bf16* Wp1 = (__bf16*)d_ws;                    // 384*256 bf16
  __bf16* Wp2 = Wp1 + (size_t)HS * KT1 * 32;      // 256*256 bf16
  float* stats = (float*)(Wp2 + (size_t)HS * KT2 * 32);
  float* gsum1 = stats;          float* gsq1 = stats + 256;
  float* gsum2 = stats + 512;    float* gsq2 = stats + 768;
  float* scale1 = stats + 1024;  float* shift1 = stats + 1280;
  float* scale2 = stats + 1536;  float* shift2 = stats + 1792;

  pack_weights_k<<<(HS * KT1 * 32 + 255) / 256, 256, 0, stream>>>(W1, W2, Wp1, Wp2, stats);

  const int nblk = (E + 63) / 64;
  gemm1_k<<<nblk, 256, 0, stream>>>(srcp, dstp, eap, up, batch, Wp1, b1,
                                    out, gsum1, gsq1, E);
  stats_k<<<1, 256, 0, stream>>>(gsum1, gsq1, g1, be1, scale1, shift1, 1.0f / (float)E);
  gemm2_k<<<nblk, 256, 0, stream>>>(out, Wp2, b2, scale1, shift1, gsum2, gsq2, E);
  stats_k<<<1, 256, 0, stream>>>(gsum2, gsq2, g2, be2, scale2, shift2, 1.0f / (float)E);

  size_t total = (size_t)E * HS;
  bn2_k<<<(unsigned)((total / 4 + 255) / 256), 256, 0, stream>>>(out, scale2, shift2, total);
}
